// _PointNet2SAModuleBase_5007931867444
// MI455X (gfx1250) — compile-verified, matrix operand order checked by reading, not run on hardware
//
#include <hip/hip_runtime.h>
#include <hip/hip_bf16.h>

// ---------------------------------------------------------------------------
// PointNet++ SA module for MI455X (gfx1250), wave32 + WMMA f16 (f32 accum).
//   B=4, N=8192, CIN=64, NPOINT=2048, NSAMPLE=32, RADIUS=0.2
//   out = [ new_xyz (4,2048,3) | new_features (4,128,2048) ] as f32, flat.
// ---------------------------------------------------------------------------

#define NB      4
#define NPTS    8192
#define NCH     64
#define NPOINT  2048
#define NSAMP   32
#define RAD2    0.04f
#define C0      67      // 3 + 64
#define C0P     96      // padded to 3 * 32 for K of wmma f16
#define H1C     64
#define H3C     128

typedef __attribute__((ext_vector_type(16))) _Float16 v16h;
typedef __attribute__((ext_vector_type(8)))  float    v8f;

// ---------------------------------------------------------------------------
// Kernel 1: furthest point sampling, one block per batch.
// Points + running min-dist live entirely in VGPRs (32 pts / thread).
// Writes new_xyz directly (coords of each selected centroid).
// ---------------------------------------------------------------------------
__global__ __launch_bounds__(256)
void fps_kernel(const float* __restrict__ xyz, float* __restrict__ newxyz) {
    const int b   = blockIdx.x;
    const int tid = threadIdx.x;               // 0..255
    const float* xb = xyz + (size_t)b * NPTS * 3;

    float px[32], py[32], pz[32], md[32];
#pragma unroll
    for (int k = 0; k < 32; ++k) {
        const int i = k * 256 + tid;
        px[k] = xb[i * 3 + 0];
        py[k] = xb[i * 3 + 1];
        pz[k] = xb[i * 3 + 2];
        md[k] = 3.0e38f;
    }

    __shared__ float redv[256];
    __shared__ int   redi[256];

    int last = 0;
    for (int it = 0; it < NPOINT; ++it) {
        // centroid coords (same address for all lanes -> broadcast load)
        const float cx = xb[last * 3 + 0];
        const float cy = xb[last * 3 + 1];
        const float cz = xb[last * 3 + 2];
        if (tid == 0) {
            float* o = newxyz + ((size_t)b * NPOINT + it) * 3;
            o[0] = cx; o[1] = cy; o[2] = cz;
        }

        float bv = -1.0f; int bi = 0;
#pragma unroll
        for (int k = 0; k < 32; ++k) {
            const float dx = px[k] - cx, dy = py[k] - cy, dz = pz[k] - cz;
            const float d  = dx * dx + dy * dy + dz * dz;
            md[k] = fminf(md[k], d);
            if (md[k] > bv) { bv = md[k]; bi = k * 256 + tid; }
        }
        redv[tid] = bv; redi[tid] = bi;
        __syncthreads();
#pragma unroll
        for (int off = 128; off > 0; off >>= 1) {
            if (tid < off) {
                const float ov = redv[tid + off];
                const int   oi = redi[tid + off];
                if (ov > redv[tid] || (ov == redv[tid] && oi < redi[tid])) {
                    redv[tid] = ov; redi[tid] = oi;
                }
            }
            __syncthreads();
        }
        last = redi[0];
        __syncthreads();   // protect red* reuse next iteration
    }
}

// ---------------------------------------------------------------------------
// WMMA fragment helpers — layouts per CDNA5 ISA 7.12.2 (wave32).
//
// 16-bit A (16x32, MxK): lane&15 -> M; lane>>4 selects K-halves:
//   VGPR v in 0..3 : K = ks*32 + 8*half + 2v   (+1)
//   VGPR v in 4..7 : K = ks*32 + 16 + 8*half + 2(v-4) (+1)
// B (KxN) mirrors with lane&15 -> N, reading K-contiguous rows (weight rows).
// ---------------------------------------------------------------------------
__device__ __forceinline__
v16h load_frag(const _Float16* __restrict__ buf, int rowStrideH,
               int tile16, int ks, int lane) {
    union { v16h h; unsigned u[8]; } r;
    const unsigned* bu = (const unsigned*)buf;
    const int rs   = rowStrideH >> 1;           // stride in dword pairs
    const int row  = tile16 * 16 + (lane & 15);
    const int half = (lane >> 4) & 1;
    const unsigned* p = bu + row * rs + ks * 16 + half * 4;
#pragma unroll
    for (int v = 0; v < 4; ++v) r.u[v] = p[v];
#pragma unroll
    for (int v = 0; v < 4; ++v) r.u[4 + v] = p[8 + v];
    return r.h;
}

// C/D (16x16 f32): VGPR r, lane -> M = mt*16 + 8*(lane>>4) + r, N = nt*16 + (lane&15)
__device__ __forceinline__
void store_frag_relu_f16(_Float16* __restrict__ buf, int rowStrideH,
                         int mt, int nt, int lane, v8f d) {
    const int mrow = mt * 16 + ((lane >> 4) & 1) * 8;
    const int ncol = nt * 16 + (lane & 15);
#pragma unroll
    for (int r = 0; r < 8; ++r) {
        float x = d[r];
        x = x > 0.0f ? x : 0.0f;
        buf[(mrow + r) * rowStrideH + ncol] = (_Float16)x;
    }
}

// ---------------------------------------------------------------------------
// Kernel 2: fused ball-query + gather + 3-layer MLP (WMMA) + max-pool.
// 64 threads = 2 waves per block; one wave per sampled center.
// ---------------------------------------------------------------------------
__global__ __launch_bounds__(64)
void sa_mlp_kernel(const float* __restrict__ xyz,
                   const float* __restrict__ feat,
                   const float* __restrict__ W1, const float* __restrict__ B1,
                   const float* __restrict__ W2, const float* __restrict__ B2,
                   const float* __restrict__ W3, const float* __restrict__ B3,
                   const float* __restrict__ newxyz,
                   float* __restrict__ outfeat) {
    __shared__ __align__(16) _Float16 sW1[64 * C0P];   // 12 KB (K padded, zeros)
    __shared__ __align__(16) _Float16 sW2[64 * 64];    //  8 KB
    __shared__ __align__(16) _Float16 sW3[128 * 64];   // 16 KB
    __shared__ float sB1[64], sB2[64], sB3[128];
    __shared__ __align__(16) _Float16 bufA[2][32 * C0P];  // G, then h2 (stride 64)
    __shared__ __align__(16) _Float16 bufB[2][32 * 64];   // h1
    __shared__ int blist[2][NSAMP];

    const int tid  = threadIdx.x;
    const int w    = tid >> 5;
    const int lane = tid & 31;

    // ---- stage weights (f16) + biases once per block ----
    for (int i = tid; i < 64 * C0P; i += 64) {
        const int o = i / C0P, c = i % C0P;
        sW1[i] = (c < C0) ? (_Float16)W1[o * C0 + c] : (_Float16)0.0f;
    }
    for (int i = tid; i < 64 * 64;  i += 64) sW2[i] = (_Float16)W2[i];
    for (int i = tid; i < 128 * 64; i += 64) sW3[i] = (_Float16)W3[i];
    for (int i = tid; i < 64;  i += 64) { sB1[i] = B1[i]; sB2[i] = B2[i]; }
    for (int i = tid; i < 128; i += 64) sB3[i] = B3[i];
    __syncthreads();

    const int ctr = blockIdx.x * 2 + w;       // global center id
    const int b   = ctr >> 11;                // / 2048
    const int s   = ctr & 2047;

    const float* xb = xyz + (size_t)b * NPTS * 3;
    const float qx = newxyz[((size_t)b * NPOINT + s) * 3 + 0];
    const float qy = newxyz[((size_t)b * NPOINT + s) * 3 + 1];
    const float qz = newxyz[((size_t)b * NPOINT + s) * 3 + 2];

    // ---- ball query: first NSAMP indices (ascending) within radius ----
    int cnt = 0;
    for (int base = 0; base < NPTS && cnt < NSAMP; base += 32) {
        const int i = base + lane;
        const float dx = xb[i * 3 + 0] - qx;
        const float dy = xb[i * 3 + 1] - qy;
        const float dz = xb[i * 3 + 2] - qz;
        const bool hit = (dx * dx + dy * dy + dz * dz) < RAD2;
        const unsigned mask = (unsigned)__ballot(hit);
        if (hit) {
            const int pos = cnt + __popc(mask & ((1u << lane) - 1u));
            if (pos < NSAMP) blist[w][pos] = i;
        }
        cnt += __popc(mask);
    }
    if (cnt > NSAMP) cnt = NSAMP;
    if (cnt < 1)     cnt = 1;                 // center itself always hits
    asm volatile("s_wait_dscnt 0x0" ::: "memory");

    // ---- gather: lane k builds row k of G = [xyz - q | feat], padded ----
    {
        const int j = blist[w][lane < cnt ? lane : 0];
        _Float16* ga = &bufA[w][lane * C0P];
        ga[0] = (_Float16)(xb[j * 3 + 0] - qx);
        ga[1] = (_Float16)(xb[j * 3 + 1] - qy);
        ga[2] = (_Float16)(xb[j * 3 + 2] - qz);
        const float* fb = feat + (size_t)b * NCH * NPTS;
        for (int c = 0; c < NCH; ++c)
            ga[3 + c] = (_Float16)fb[(size_t)c * NPTS + j];
#pragma unroll
        for (int c = C0; c < C0P; ++c) ga[c] = (_Float16)0.0f;
    }
    asm volatile("s_wait_dscnt 0x0" ::: "memory");

    // ---- layer 1: (32xC0P) x (C0Px64) -> h1 in bufB ----
#pragma unroll
    for (int mt = 0; mt < 2; ++mt) {
#pragma unroll
        for (int nt = 0; nt < 4; ++nt) {
            const float bias = sB1[nt * 16 + (lane & 15)];
            v8f acc = {bias, bias, bias, bias, bias, bias, bias, bias};
#pragma unroll
            for (int ks = 0; ks < 3; ++ks) {
                v16h a  = load_frag(bufA[w], C0P, mt, ks, lane);
                v16h bm = load_frag(sW1,    C0P, nt, ks, lane);
                acc = __builtin_amdgcn_wmma_f32_16x16x32_f16(
                          false, a, false, bm, (short)0, acc, false, false);
            }
            store_frag_relu_f16(bufB[w], 64, mt, nt, lane, acc);
        }
    }
    asm volatile("s_wait_dscnt 0x0" ::: "memory");

    // ---- layer 2: (32x64) x (64x64) -> h2 in bufA (stride 64) ----
#pragma unroll
    for (int mt = 0; mt < 2; ++mt) {
#pragma unroll
        for (int nt = 0; nt < 4; ++nt) {
            const float bias = sB2[nt * 16 + (lane & 15)];
            v8f acc = {bias, bias, bias, bias, bias, bias, bias, bias};
#pragma unroll
            for (int ks = 0; ks < 2; ++ks) {
                v16h a  = load_frag(bufB[w], 64, mt, ks, lane);
                v16h bm = load_frag(sW2,    64, nt, ks, lane);
                acc = __builtin_amdgcn_wmma_f32_16x16x32_f16(
                          false, a, false, bm, (short)0, acc, false, false);
            }
            store_frag_relu_f16(bufA[w], 64, mt, nt, lane, acc);
        }
    }
    asm volatile("s_wait_dscnt 0x0" ::: "memory");

    // ---- layer 3: (32x64) x (64x128) + relu + max over 32 samples ----
#pragma unroll
    for (int nt = 0; nt < 8; ++nt) {
        const float bias = sB3[nt * 16 + (lane & 15)];
        float pmax = 0.0f;                    // relu outputs are >= 0
#pragma unroll
        for (int mt = 0; mt < 2; ++mt) {
            v8f acc = {bias, bias, bias, bias, bias, bias, bias, bias};
#pragma unroll
            for (int ks = 0; ks < 2; ++ks) {
                v16h a  = load_frag(bufA[w], 64, mt, ks, lane);
                v16h bm = load_frag(sW3,    64, nt, ks, lane);
                acc = __builtin_amdgcn_wmma_f32_16x16x32_f16(
                          false, a, false, bm, (short)0, acc, false, false);
            }
#pragma unroll
            for (int r = 0; r < 8; ++r) {
                float x = acc[r];
                x = x > 0.0f ? x : 0.0f;
                pmax = fmaxf(pmax, x);
            }
        }
        // lane l and l+16 hold the two row-halves of the same output column
        pmax = fmaxf(pmax, __shfl_xor(pmax, 16, 32));
        if (lane < 16) {
            const int o = nt * 16 + lane;
            outfeat[((size_t)b * H3C + o) * NPOINT + s] = pmax;
        }
    }
}

// ---------------------------------------------------------------------------
extern "C" void kernel_launch(void* const* d_in, const int* in_sizes, int n_in,
                              void* d_out, int out_size, void* d_ws, size_t ws_size,
                              hipStream_t stream) {
    const float* xyz  = (const float*)d_in[0];
    const float* feat = (const float*)d_in[1];
    const float* W1   = (const float*)d_in[2];
    const float* B1   = (const float*)d_in[3];
    const float* W2   = (const float*)d_in[4];
    const float* B2   = (const float*)d_in[5];
    const float* W3   = (const float*)d_in[6];
    const float* B3   = (const float*)d_in[7];

    float* out     = (float*)d_out;
    float* newxyz  = out;                                  // (4,2048,3)
    float* outfeat = out + (size_t)NB * NPOINT * 3;        // (4,128,2048)

    fps_kernel<<<NB, 256, 0, stream>>>(xyz, newxyz);
    sa_mlp_kernel<<<(NB * NPOINT) / 2, 64, 0, stream>>>(
        xyz, feat, W1, B1, W2, B2, W3, B3, newxyz, outfeat);
}
